// TransLayer_81621558493779
// MI455X (gfx1250) — compile-verified, matrix operand order checked by reading, not run on hardware
//
#include <hip/hip_runtime.h>
#include <math.h>

// ---------------- types ----------------
typedef __bf16 bf16;
typedef float  v8f   __attribute__((ext_vector_type(8)));
typedef float  v2f   __attribute__((ext_vector_type(2)));
typedef bf16   v16bf __attribute__((ext_vector_type(16)));
typedef bf16   v8bf  __attribute__((ext_vector_type(8)));
typedef int    v4i   __attribute__((ext_vector_type(4)));

// ---------------- problem constants ----------------
#define BATCH   4
#define NSEQ    4096
#define DMODEL  512
#define HEADS   8
#define DH      64
#define NL      256                    // landmarks
#define BH      (BATCH*HEADS)          // 32
#define NTOK    (BATCH*NSEQ)           // 16384
#define QSCALE  0.125f                 // 64^-0.5
#define KW      33
#define KPAD    16

// =====================================================================
// Async global->LDS copy (CDNA5), guarded with a synchronous fallback
// =====================================================================
#if __has_builtin(__builtin_amdgcn_global_load_async_to_lds_b128)
#define HAVE_ASYNC 1
typedef __attribute__((address_space(1))) v4i* as1_v4i;
typedef __attribute__((address_space(3))) v4i* as3_v4i;
__device__ __forceinline__ void async_copy16(const bf16* g, bf16* l) {
  __builtin_amdgcn_global_load_async_to_lds_b128(
      (as1_v4i)(void*)g, (as3_v4i)(void*)l, 0, 0);
}
__device__ __forceinline__ void async_wait() {
#if __has_builtin(__builtin_amdgcn_s_wait_asynccnt)
  __builtin_amdgcn_s_wait_asynccnt(0);
#else
  asm volatile("s_wait_asynccnt 0" ::: "memory");
#endif
}
#else
#define HAVE_ASYNC 0
__device__ __forceinline__ void async_copy16(const bf16* g, bf16* l) {
  *(v8bf*)l = *(const v8bf*)g;
}
__device__ __forceinline__ void async_wait() {}
#endif

// =====================================================================
// WMMA fragment loader (CDNA5 wave32 16-bit layout, ISA 7.12.2).
// Source must be row-major-for-the-fragment: lane l holds row (M or N) = l,
// K runs contiguously -> two 16-byte loads per fragment.
// =====================================================================
__device__ __forceinline__ v16bf frag_rm(const bf16* p, int ld, int row0, int k0) {
  const int lane = threadIdx.x & 31, l = lane & 15, hi = lane >> 4;
  const bf16* r = p + (size_t)(row0 + l) * ld + k0 + hi * 8;
  const v8bf lo = *(const v8bf*)r;
  const v8bf hh = *(const v8bf*)(r + 16);
  return __builtin_shufflevector(lo, hh, 0, 1, 2, 3, 4, 5, 6, 7,
                                 8, 9, 10, 11, 12, 13, 14, 15);
}

#define WMMA_BF16(a, b, c) \
  __builtin_amdgcn_wmma_f32_16x16x32_bf16(false, (a), false, (b), (short)0, (c), false, false)
#define WMMA_F32(a, b, c) \
  __builtin_amdgcn_wmma_f32_16x16x4_f32(false, (a), false, (b), (short)0, (c), false, false)

// =====================================================================
// LayerNorm -> bf16  (one wave per 512-wide row)
// =====================================================================
__global__ __launch_bounds__(256) void k_layernorm(
    const float* __restrict__ x, const float* __restrict__ w,
    const float* __restrict__ b, bf16* __restrict__ h) {
  const int wave = threadIdx.x >> 5, lane = threadIdx.x & 31;
  const size_t row = (size_t)blockIdx.x * 8 + wave;
  const float* xr = x + row * DMODEL;
  float vals[16], s = 0.f, s2 = 0.f;
#pragma unroll
  for (int i = 0; i < 16; ++i) { float v = xr[lane + 32 * i]; vals[i] = v; s += v; s2 += v * v; }
#pragma unroll
  for (int off = 1; off < 32; off <<= 1) { s += __shfl_xor(s, off, 32); s2 += __shfl_xor(s2, off, 32); }
  const float mu = s * (1.f / DMODEL);
  const float rs = rsqrtf(s2 * (1.f / DMODEL) - mu * mu + 1e-5f);
  bf16* hr = h + row * DMODEL;
#pragma unroll
  for (int i = 0; i < 16; ++i) {
    const int c = lane + 32 * i;
    hr[c] = (bf16)((vals[i] - mu) * rs * w[c] + b[c]);
  }
}

// ---------------- fp32 -> bf16 (flat) ----------------
__global__ __launch_bounds__(256) void k_cvt(const float* __restrict__ s, bf16* __restrict__ d, size_t n) {
  const size_t i = (size_t)blockIdx.x * 256 + threadIdx.x;
  if (i < n) d[i] = (bf16)s[i];
}
// ---------------- fp32 [K][N] -> bf16 transposed [N][K] ----------------
__global__ __launch_bounds__(256) void k_cvtw(const float* __restrict__ s, bf16* __restrict__ d,
                                              int K_, int N_) {
  const size_t idx = (size_t)blockIdx.x * 256 + threadIdx.x;  // N_*K_
  const int n = (int)(idx / K_), k = (int)(idx % K_);
  d[idx] = (bf16)s[(size_t)k * N_ + n];
}
// ---------------- t2f [bh][m][d] fp32 -> t2bT [bh][d][m] bf16 ----------------
__global__ __launch_bounds__(256) void k_cvt_t2(const float* __restrict__ s, bf16* __restrict__ d) {
  const size_t idx = (size_t)blockIdx.x * 256 + threadIdx.x;  // BH*DH*NL
  const size_t bh = idx / (DH * NL);
  const int rem = (int)(idx % (DH * NL));
  const int dd = rem / NL, m = rem % NL;
  d[idx] = (bf16)s[(bh * NL + m) * DH + dd];
}

// =====================================================================
// QKV GEMM: (16384x512 bf16) @ Wt(1536x512 bf16, pre-transposed).
// A-tile staged through LDS with async copies; q/k head-major, v transposed.
// =====================================================================
__global__ __launch_bounds__(256) void k_gemm_qkv(
    const bf16* __restrict__ A, const bf16* __restrict__ Wt,
    bf16* __restrict__ Q, bf16* __restrict__ K, bf16* __restrict__ VT) {
  const int wave = threadIdx.x >> 5, wr = wave & 1, wc = wave >> 1;
  const int rowB = blockIdx.x * 32;
  const int col0 = blockIdx.y * 64 + wc * 16;
  __shared__ alignas(16) bf16 Atile[32 * 32];
  v8f acc = {};
  for (int kk = 0; kk < DMODEL; kk += 32) {
    if (threadIdx.x < 128) {
      const int rr = threadIdx.x >> 2, cc = (threadIdx.x & 3) * 8;
      async_copy16(A + (size_t)(rowB + rr) * DMODEL + kk + cc, &Atile[rr * 32 + cc]);
    }
    async_wait();
    __syncthreads();
    const v16bf a = frag_rm(Atile, 32, wr * 16, 0);
    const v16bf b = frag_rm(Wt, DMODEL, col0, kk);
    acc = WMMA_BF16(a, b, acc);
    __syncthreads();
  }
  const int lane = threadIdx.x & 31, l = lane & 15, hi = lane >> 4;
#pragma unroll
  for (int r = 0; r < 8; ++r) {
    const int t = rowB + wr * 16 + r + hi * 8;
    const int c = col0 + l;
    const int part = c >> 9, rem = c & 511;
    const int head = rem >> 6, d = rem & 63;
    const int bb = t >> 12, i = t & (NSEQ - 1);
    float v = acc[r];
    if (part == 0) v *= QSCALE;
    if (part == 2) {
      VT[(((size_t)(bb * HEADS + head)) * DH + d) * NSEQ + i] = (bf16)v;
    } else {
      bf16* dst = (part == 0) ? Q : K;
      dst[(((size_t)(bb * HEADS + head)) * NSEQ + i) * DH + d] = (bf16)v;
    }
  }
}

// ---------------- landmarks: mean over 16 consecutive tokens ----------------
__global__ __launch_bounds__(256) void k_landmark(const bf16* __restrict__ S, bf16* __restrict__ Lm) {
  const size_t idx = (size_t)blockIdx.x * 256 + threadIdx.x;  // BH*256*64
  const int d = idx & 63;
  const size_t r = idx >> 6;
  const int m = (int)(r & 255);
  const size_t bh = r >> 8;
  const bf16* src = S + (bh * NSEQ + (size_t)m * 16) * DH + d;
  float s = 0.f;
#pragma unroll
  for (int j = 0; j < 16; ++j) s += (float)src[(size_t)j * DH];
  Lm[idx] = (bf16)(s * (1.f / 16.f));
}

// =====================================================================
// sim2 = q_l @ k_l^T  (batched 256x256, K=64, bf16 WMMA, fp32 out)
// =====================================================================
__global__ __launch_bounds__(256) void k_gemm_qkT(
    const bf16* __restrict__ A, const bf16* __restrict__ B, float* __restrict__ C,
    int N_, long sA, long sB, long sC) {
  const int bh = blockIdx.z;
  const bf16* a = A + (size_t)bh * sA;
  const bf16* b = B + (size_t)bh * sB;
  float* c = C + (size_t)bh * sC;
  const int wave = threadIdx.x >> 5, wr = wave & 1, wc = wave >> 1;
  const int row0 = blockIdx.x * 32 + wr * 16;
  const int col0 = blockIdx.y * 64 + wc * 16;
  v8f acc = {};
  acc = WMMA_BF16(frag_rm(a, DH, row0, 0), frag_rm(b, DH, col0, 0), acc);
  acc = WMMA_BF16(frag_rm(a, DH, row0, 32), frag_rm(b, DH, col0, 32), acc);
  const int lane = threadIdx.x & 31, l = lane & 15, hi = lane >> 4;
#pragma unroll
  for (int r = 0; r < 8; ++r)
    c[(size_t)(row0 + r + hi * 8) * N_ + col0 + l] = acc[r];
}

// ---------------- row softmax, length 256 ----------------
__global__ __launch_bounds__(256) void k_softmax256(float* __restrict__ X) {
  const size_t row = blockIdx.x;
  float* xr = X + row * 256;
  const int lane = threadIdx.x & 31, wave = threadIdx.x >> 5;
  __shared__ float red[8];
  float v = xr[threadIdx.x];
  float m = v;
#pragma unroll
  for (int off = 1; off < 32; off <<= 1) m = fmaxf(m, __shfl_xor(m, off, 32));
  if (lane == 0) red[wave] = m;
  __syncthreads();
  float gm = red[0];
#pragma unroll
  for (int j = 1; j < 8; ++j) gm = fmaxf(gm, red[j]);
  __syncthreads();
  const float e = __expf(v - gm);
  float s = e;
#pragma unroll
  for (int off = 1; off < 32; off <<= 1) s += __shfl_xor(s, off, 32);
  if (lane == 0) red[wave] = s;
  __syncthreads();
  float gs = 0.f;
#pragma unroll
  for (int j = 0; j < 8; ++j) gs += red[j];
  xr[threadIdx.x] = e / gs;
}

// ---------------- pinv init reductions ----------------
__global__ void k_redinit(int* rb) { if (threadIdx.x == 0) { rb[0] = 0; rb[1] = 0; } }

__global__ __launch_bounds__(32) void k_rowabs(const float* __restrict__ X, int* rb) {
  const size_t row = blockIdx.x;  // BH*256
  const float* xr = X + row * 256;
  float s = 0.f;
  for (int j = threadIdx.x; j < 256; j += 32) s += fabsf(xr[j]);
#pragma unroll
  for (int off = 1; off < 32; off <<= 1) s += __shfl_xor(s, off, 32);
  if (threadIdx.x == 0) atomicMax(rb + 0, __float_as_int(s));
}
__global__ __launch_bounds__(32) void k_colabs(const float* __restrict__ X, int* rb) {
  const int bh = blockIdx.x >> 8, col = blockIdx.x & 255;
  const float* xb = X + (size_t)bh * 65536 + col;
  float s = 0.f;
  for (int i = threadIdx.x; i < 256; i += 32) s += fabsf(xb[(size_t)i * 256]);
#pragma unroll
  for (int off = 1; off < 32; off <<= 1) s += __shfl_xor(s, off, 32);
  if (threadIdx.x == 0) atomicMax(rb + 1, __float_as_int(s));
}
__global__ __launch_bounds__(256) void k_initz(const float* __restrict__ X, float* __restrict__ Z,
                                               const int* __restrict__ rb) {
  const size_t idx = (size_t)blockIdx.x * 256 + threadIdx.x;  // BH*256*256
  const size_t bh = idx >> 16;
  const int i = (int)((idx >> 8) & 255), j = (int)(idx & 255);
  const float denom = __int_as_float(rb[0]) * __int_as_float(rb[1]);
  Z[(bh << 16) + (size_t)i * 256 + j] = X[(bh << 16) + (size_t)j * 256 + i] / denom;
}
__global__ __launch_bounds__(256) void k_axpyI(const float* __restrict__ P, float* __restrict__ U,
                                               float alpha) {
  const size_t idx = (size_t)blockIdx.x * 256 + threadIdx.x;
  const int i = (int)((idx >> 8) & 255), j = (int)(idx & 255);
  U[idx] = ((i == j) ? alpha : 0.f) - P[idx];
}

// =====================================================================
// Batched fp32 GEMM via V_WMMA_F32_16X16X4_F32 (pinv chain)
// mode 0: C=A@B ; mode 2: C=alpha*(A@B)
// =====================================================================
__global__ __launch_bounds__(256) void k_gemm_f32(
    const float* __restrict__ A, const float* __restrict__ B, float* __restrict__ C,
    int N_, int K_, long sA, long sB, long sC, int mode, float alpha) {
  const int bh = blockIdx.z;
  const float* a = A + (size_t)bh * sA;
  const float* b = B + (size_t)bh * sB;
  float* c = C + (size_t)bh * sC;
  const int wave = threadIdx.x >> 5, wr = wave & 1, wc = wave >> 1;
  const int row0 = blockIdx.x * 32 + wr * 16;
  const int col0 = blockIdx.y * 64 + wc * 16;
  const int lane = threadIdx.x & 31, l = lane & 15, hi = lane >> 4;
  const float* ar = a + (size_t)(row0 + l) * K_ + hi * 2;
  const float* bc = b + col0 + l + (size_t)(hi * 2) * N_;
  v8f acc = {};
#pragma unroll 4
  for (int kk = 0; kk < K_; kk += 4) {
    const v2f av = *(const v2f*)(ar + kk);
    v2f bv;
    bv[0] = bc[(size_t)kk * N_];
    bv[1] = bc[(size_t)(kk + 1) * N_];
    acc = WMMA_F32(av, bv, acc);
  }
#pragma unroll
  for (int r = 0; r < 8; ++r) {
    float v = acc[r];
    if (mode == 2) v *= alpha;
    c[(size_t)(row0 + r + hi * 8) * N_ + col0 + l] = v;
  }
}

// =====================================================================
// Flash-style softmax(Q K^T) @ V — one wave per 16 query rows.
// Q: [nq x 64] bf16, K: [nkv x 64] bf16, VT: [64 x nkv] bf16 (ld=ldv),
// O fp32 [nq x 64]. nkv streamed in 32-chunks; p-tile re-layout via LDS.
// =====================================================================
__global__ __launch_bounds__(32) void k_flash(
    const bf16* __restrict__ Qg, const bf16* __restrict__ Kg,
    const bf16* __restrict__ Vt, float* __restrict__ Og,
    int nkv, int ldv, long sQ, long sK, long sV, long sO) {
  const int bh = blockIdx.y;
  const bf16* q = Qg + (size_t)bh * sQ;
  const bf16* k = Kg + (size_t)bh * sK;
  const bf16* v = Vt + (size_t)bh * sV;
  float* o = Og + (size_t)bh * sO;
  const int m0 = blockIdx.x * 16;
  const int lane = threadIdx.x, l = lane & 15, hi = lane >> 4;

  const v16bf qa0 = frag_rm(q, DH, m0, 0);
  const v16bf qa1 = frag_rm(q, DH, m0, 32);

  v8f ot[4] = {};
  float mrow[8], srow[8];
#pragma unroll
  for (int r = 0; r < 8; ++r) { mrow[r] = -3.0e38f; srow[r] = 0.f; }

  __shared__ alignas(16) bf16 pL[16 * 32];

  for (int n0 = 0; n0 < nkv; n0 += 32) {
    v8f s0 = {}, s1 = {};
    s0 = WMMA_BF16(qa0, frag_rm(k, DH, n0, 0), s0);
    s0 = WMMA_BF16(qa1, frag_rm(k, DH, n0, 32), s0);
    s1 = WMMA_BF16(qa0, frag_rm(k, DH, n0 + 16, 0), s1);
    s1 = WMMA_BF16(qa1, frag_rm(k, DH, n0 + 16, 32), s1);

#pragma unroll
    for (int r = 0; r < 8; ++r) {
      float mx = fmaxf(s0[r], s1[r]);
#pragma unroll
      for (int off = 1; off < 16; off <<= 1) mx = fmaxf(mx, __shfl_xor(mx, off, 32));
      const float mnew = fmaxf(mrow[r], mx);
      const float scale = __expf(mrow[r] - mnew);
      const float p0 = __expf(s0[r] - mnew);
      const float p1 = __expf(s1[r] - mnew);
      float ps = p0 + p1;
#pragma unroll
      for (int off = 1; off < 16; off <<= 1) ps += __shfl_xor(ps, off, 32);
      srow[r] = srow[r] * scale + ps;
      mrow[r] = mnew;
#pragma unroll
      for (int t = 0; t < 4; ++t) ot[t][r] *= scale;
      pL[(r + hi * 8) * 32 + l] = (bf16)p0;
      pL[(r + hi * 8) * 32 + 16 + l] = (bf16)p1;
    }
    __syncthreads();
    const v16bf pf = frag_rm((const bf16*)pL, 32, 0, 0);
#pragma unroll
    for (int t = 0; t < 4; ++t)
      ot[t] = WMMA_BF16(pf, frag_rm(v, ldv, t * 16, n0), ot[t]);
    __syncthreads();
  }
#pragma unroll
  for (int r = 0; r < 8; ++r) {
    const float inv = 1.f / srow[r];
    float* orow = o + (size_t)(m0 + r + hi * 8) * DH;
#pragma unroll
    for (int t = 0; t < 4; ++t) orow[t * 16 + l] = ot[t][r] * inv;
  }
}

// ---------------- depthwise conv residual (VT layout: [bh][d][n]) ----------------
__global__ __launch_bounds__(256) void k_convres(
    const bf16* __restrict__ VT, const float* __restrict__ wres, float* __restrict__ O) {
  const size_t idx = (size_t)blockIdx.x * 256 + threadIdx.x;  // BH*NSEQ*DH, O layout [bh][n][d]
  const int d = (int)(idx & 63);
  const size_t t = idx >> 6;
  const int i = (int)(t & (NSEQ - 1));
  const size_t bhh = t >> 12;
  const int h = (int)(bhh & (HEADS - 1));
  const bf16* vb = VT + (bhh * DH + d) * NSEQ;
  float acc = 0.f;
#pragma unroll 1
  for (int tt = 0; tt < KW; ++tt) {
    const int src = i + tt - KPAD;
    if (src >= 0 && src < NSEQ) acc += wres[h * KW + tt] * (float)vb[src];
  }
  O[idx] += acc;
}

// ---------------- [b,h,n,d] fp32 -> [b,n,h*d] bf16 ----------------
__global__ __launch_bounds__(256) void k_headmerge(const float* __restrict__ Oh, bf16* __restrict__ Ao) {
  const size_t idx = (size_t)blockIdx.x * 256 + threadIdx.x;  // NTOK*512
  const int c = (int)(idx & 511);
  const size_t t = idx >> 9;
  const int b = (int)(t >> 12), i = (int)(t & (NSEQ - 1));
  const int h = c >> 6, d = c & 63;
  Ao[idx] = (bf16)Oh[(((size_t)(b * HEADS + h)) * NSEQ + i) * DH + d];
}

// =====================================================================
// Output GEMM: y = x + Ao @ W_out + b_out   (W_out pre-transposed [N][K])
// =====================================================================
__global__ __launch_bounds__(256) void k_gemm_out(
    const bf16* __restrict__ A, const bf16* __restrict__ Wt,
    const float* __restrict__ X, const float* __restrict__ bias, float* __restrict__ Y) {
  const int wave = threadIdx.x >> 5, wr = wave & 1, wc = wave >> 1;
  const int row0 = blockIdx.x * 32 + wr * 16;
  const int col0 = blockIdx.y * 64 + wc * 16;
  v8f acc = {};
  for (int kk = 0; kk < DMODEL; kk += 32)
    acc = WMMA_BF16(frag_rm(A, DMODEL, row0, kk), frag_rm(Wt, DMODEL, col0, kk), acc);
  const int lane = threadIdx.x & 31, l = lane & 15, hi = lane >> 4;
#pragma unroll
  for (int r = 0; r < 8; ++r) {
    const size_t t = row0 + r + hi * 8;
    const int c = col0 + l;
    Y[t * DMODEL + c] = X[t * DMODEL + c] + acc[r] + bias[c];
  }
}

// =====================================================================
// host-side launch
// =====================================================================
extern "C" void kernel_launch(void* const* d_in, const int* in_sizes, int n_in,
                              void* d_out, int out_size, void* d_ws, size_t ws_size,
                              hipStream_t stream) {
  const float* x     = (const float*)d_in[0];
  const float* ln_w  = (const float*)d_in[1];
  const float* ln_b  = (const float*)d_in[2];
  const float* w_qkv = (const float*)d_in[3];
  const float* w_out = (const float*)d_in[4];
  const float* b_out = (const float*)d_in[5];
  const float* w_res = (const float*)d_in[6];
  float* y = (float*)d_out;

  char* ws = (char*)d_ws;
  size_t off = 0;
  auto alloc = [&](size_t bytes) { char* p = ws + off; off = (off + bytes + 255) & ~(size_t)255; return p; };

  bf16*  h_bf   = (bf16*)alloc((size_t)NTOK * DMODEL * 2);
  bf16*  wqkv_t = (bf16*)alloc((size_t)DMODEL * 3 * DMODEL * 2);  // [1536][512]
  bf16*  wout_t = (bf16*)alloc((size_t)DMODEL * DMODEL * 2);      // [512][512]
  bf16*  q_bf   = (bf16*)alloc((size_t)BH * NSEQ * DH * 2);       // [bh][n][d]
  bf16*  k_bf   = (bf16*)alloc((size_t)BH * NSEQ * DH * 2);       // [bh][n][d]
  bf16*  v_t    = (bf16*)alloc((size_t)BH * NSEQ * DH * 2);       // [bh][d][n]
  bf16*  ql_bf  = (bf16*)alloc((size_t)BH * NL * DH * 2);
  bf16*  kl_bf  = (bf16*)alloc((size_t)BH * NL * DH * 2);
  float* att2   = (float*)alloc((size_t)BH * NL * NL * 4);
  float* z_a    = (float*)alloc((size_t)BH * NL * NL * 4);
  float* z_b    = (float*)alloc((size_t)BH * NL * NL * 4);
  float* xz     = (float*)alloc((size_t)BH * NL * NL * 4);
  float* ubuf   = (float*)alloc((size_t)BH * NL * NL * 4);
  float* tbuf   = (float*)alloc((size_t)BH * NL * NL * 4);
  float* w3v    = (float*)alloc((size_t)BH * NL * DH * 4);
  float* t2f    = (float*)alloc((size_t)BH * NL * DH * 4);
  bf16*  t2bt   = (bf16*)alloc((size_t)BH * DH * NL * 2);         // [bh][d][m]
  float* outh   = (float*)alloc((size_t)BH * NSEQ * DH * 4);
  bf16*  ao_bf  = (bf16*)alloc((size_t)NTOK * DMODEL * 2);
  int*   redbuf = (int*)alloc(256);

  // 1) LayerNorm + bf16
  k_layernorm<<<NTOK / 8, 256, 0, stream>>>(x, ln_w, ln_b, h_bf);
  // 2) weight conversion (transposed to [N][K] bf16)
  k_cvtw<<<(3 * DMODEL * DMODEL) / 256, 256, 0, stream>>>(w_qkv, wqkv_t, DMODEL, 3 * DMODEL);
  k_cvtw<<<(DMODEL * DMODEL) / 256, 256, 0, stream>>>(w_out, wout_t, DMODEL, DMODEL);
  // 3) QKV projection (bf16 WMMA, async LDS staging)
  k_gemm_qkv<<<dim3(NTOK / 32, (3 * DMODEL) / 64), 256, 0, stream>>>(h_bf, wqkv_t, q_bf, k_bf, v_t);
  // 4) landmarks
  k_landmark<<<(BH * NL * DH) / 256, 256, 0, stream>>>(q_bf, ql_bf);
  k_landmark<<<(BH * NL * DH) / 256, 256, 0, stream>>>(k_bf, kl_bf);
  // 5) attn2 = softmax(q_l k_l^T)
  k_gemm_qkT<<<dim3(NL / 32, NL / 64, BH), 256, 0, stream>>>(
      ql_bf, kl_bf, att2, NL, (long)NL * DH, (long)NL * DH, (long)NL * NL);
  k_softmax256<<<BH * NL, 256, 0, stream>>>(att2);
  // 6) pinv init: z = attn2^T / (colmax * rowmax)
  k_redinit<<<1, 32, 0, stream>>>(redbuf);
  k_rowabs<<<BH * NL, 32, 0, stream>>>(att2, redbuf);
  k_colabs<<<BH * NL, 32, 0, stream>>>(att2, redbuf);
  k_initz<<<(BH * NL * NL) / 256, 256, 0, stream>>>(att2, z_a, redbuf);
  // 7) 6 Newton–Schulz iterations (fp32 WMMA)
  float* zc = z_a; float* zn = z_b;
  const long sM = (long)NL * NL;
  for (int it = 0; it < 6; ++it) {
    k_gemm_f32<<<dim3(NL / 32, NL / 64, BH), 256, 0, stream>>>(att2, zc, xz, NL, NL, sM, sM, sM, 0, 0.f);
    k_axpyI<<<(BH * NL * NL) / 256, 256, 0, stream>>>(xz, ubuf, 7.f);
    k_gemm_f32<<<dim3(NL / 32, NL / 64, BH), 256, 0, stream>>>(xz, ubuf, tbuf, NL, NL, sM, sM, sM, 0, 0.f);
    k_axpyI<<<(BH * NL * NL) / 256, 256, 0, stream>>>(tbuf, ubuf, 15.f);
    k_gemm_f32<<<dim3(NL / 32, NL / 64, BH), 256, 0, stream>>>(xz, ubuf, tbuf, NL, NL, sM, sM, sM, 0, 0.f);
    k_axpyI<<<(BH * NL * NL) / 256, 256, 0, stream>>>(tbuf, ubuf, 13.f);
    k_gemm_f32<<<dim3(NL / 32, NL / 64, BH), 256, 0, stream>>>(zc, ubuf, zn, NL, NL, sM, sM, sM, 2, 0.25f);
    float* t = zc; zc = zn; zn = t;
  }
  // 8) w3v = softmax(q_l k^T) @ v   (flash over nkv=4096, V transposed)
  k_flash<<<dim3(NL / 16, BH), 32, 0, stream>>>(
      ql_bf, k_bf, v_t, w3v, NSEQ, NSEQ,
      (long)NL * DH, (long)NSEQ * DH, (long)NSEQ * DH, (long)NL * DH);
  // 9) tmp2 = z @ w3v (fp32 WMMA, N=64) -> bf16 transposed
  k_gemm_f32<<<dim3(NL / 32, 1, BH), 256, 0, stream>>>(
      zc, w3v, t2f, DH, NL, sM, (long)NL * DH, (long)NL * DH, 0, 0.f);
  k_cvt_t2<<<(BH * NL * DH) / 256, 256, 0, stream>>>(t2f, t2bt);
  // 10) out_head = softmax(q k_l^T) @ tmp2  (flash over nkv=256)
  k_flash<<<dim3(NSEQ / 16, BH), 32, 0, stream>>>(
      q_bf, kl_bf, t2bt, outh, NL, NL,
      (long)NSEQ * DH, (long)NL * DH, (long)DH * NL, (long)NSEQ * DH);
  // 11) depthwise conv residual added in place
  k_convres<<<(BH * NSEQ * DH) / 256, 256, 0, stream>>>(v_t, w_res, outh);
  // 12) merge heads -> bf16, output projection with residual + bias
  k_headmerge<<<(NTOK * DMODEL) / 256, 256, 0, stream>>>(outh, ao_bf);
  k_gemm_out<<<dim3(NTOK / 32, DMODEL / 64), 256, 0, stream>>>(ao_bf, wout_t, x, b_out, y);
  (void)in_sizes; (void)n_in; (void)out_size; (void)ws_size; (void)off;
}
